// AxialAttention_84971632984398
// MI455X (gfx1250) — compile-verified, matrix operand order checked by reading, not run on hardware
//
#include <hip/hip_runtime.h>
#include <hip/hip_bf16.h>
#include <math.h>
#include <cstdint>

typedef __attribute__((ext_vector_type(16))) __bf16 v16bf;
typedef __attribute__((ext_vector_type(8)))  float  v8f;

#define ALPHA_C 0.08284766435f   // 1 - tanh(pi/2)
#define EPS_C   1e-5f

// ---- CDNA5 async global->LDS copy (ASYNCcnt), with portable fallback ------
#if defined(__has_builtin)
#if __has_builtin(__builtin_amdgcn_global_load_async_to_lds_b128) && \
    __has_builtin(__builtin_amdgcn_s_wait_asynccnt)
#define USE_ASYNC_LDS 1
#endif
#endif
#ifndef USE_ASYNC_LDS
#define USE_ASYNC_LDS 0
#endif

typedef int vi4 __attribute__((vector_size(16)));
typedef __attribute__((address_space(1))) vi4* as1_vi4p;
typedef __attribute__((address_space(3))) vi4* as3_vi4p;

__device__ __forceinline__ void cp16_g2l(void* lds_dst, const void* gsrc) {
#if USE_ASYNC_LDS
    __builtin_amdgcn_global_load_async_to_lds_b128(
        (as1_vi4p)(uintptr_t)gsrc,
        (as3_vi4p)(uintptr_t)lds_dst,
        0, 0);
#else
    *(uint4*)lds_dst = *(const uint4*)gsrc;
#endif
}
__device__ __forceinline__ void wait_async_lds() {
#if USE_ASYNC_LDS
    __builtin_amdgcn_s_wait_asynccnt(0);
#endif
}

__device__ __forceinline__ v8f wmma_bf16(v16bf a, v16bf b, v8f c) {
    return __builtin_amdgcn_wmma_f32_16x16x32_bf16(
        false, a, false, b, (short)0, c, false, false);
}

// A/B fragment from [row][k] bf16 LDS tile (stride S elements, S%8==0 so the
// two 8-element halves are 16B-aligned ds_load_b128).
template <int S>
__device__ __forceinline__ v16bf frag_row(const __bf16* base, int row0, int k0, int lane) {
    v16bf f;
    const int r = row0 + (lane & 15);
    const int half = lane >> 4;
#pragma unroll
    for (int i = 0; i < 8; ++i) {
        const int kk = k0 + (i >> 2) * 16 + half * 8 + (i & 3) * 2;
        f[2 * i]     = base[r * S + kk];
        f[2 * i + 1] = base[r * S + kk + 1];
    }
    return f;
}

// ---------------------------------------------------------------------------
// Pre-pass kernels (bandwidth-trivial vs the GEMMs)
// ---------------------------------------------------------------------------
__global__ __launch_bounds__(256) void cvt_bf16_kernel(const float* __restrict__ s,
                                                       __bf16* __restrict__ d) {
    const size_t i = ((size_t)blockIdx.x * 256 + threadIdx.x) * 4;
    const float4 v = *(const float4*)(s + i);
    d[i + 0] = (__bf16)v.x; d[i + 1] = (__bf16)v.y;
    d[i + 2] = (__bf16)v.z; d[i + 3] = (__bf16)v.w;
}

// w[R][C] f32 -> wt[C][R] bf16 (exact grid: R*C/256 blocks)
__global__ __launch_bounds__(256) void transpose_bf16_kernel(const float* __restrict__ w,
                                                             __bf16* __restrict__ wt, int R, int C) {
    const int o = blockIdx.x * 256 + threadIdx.x;
    const int c = o / R, r = o % R;
    wt[o] = (__bf16)w[(size_t)r * C + c];
}

__global__ __launch_bounds__(256) void zero_kernel(float* __restrict__ p, int n) {
    const int i = blockIdx.x * 256 + threadIdx.x;
    if (i < n) p[i] = 0.f;
}

// ---------------------------------------------------------------------------
// GEMM1: qkv[m][nn] = xb[m][:] . wqkvT[nn][:]  (M=25088,K=1024,N=3072)
// epilogue: += ALPHA*img, gamma on k-third. 128x128 block, BK=32,
// double-buffered LDS, async global->LDS, 8 WMMA / wave / K-step.
// ---------------------------------------------------------------------------
__global__ __launch_bounds__(256)
void gemm_qkv_kernel(const __bf16* __restrict__ Ag, const __bf16* __restrict__ Bg,
                     const float* __restrict__ img, float* __restrict__ qkv) {
    __shared__ __bf16 As[2][128][40];
    __shared__ __bf16 Bs[2][128][40];
    const int bn = blockIdx.x;        // 0..23
    const int bm = blockIdx.y;        // 0..195
    const int tid = threadIdx.x;
    const int lane = tid & 31, wave = tid >> 5;
    const int wm = wave & 3, wn = wave >> 2;

    const int r0 = tid >> 2;                 // rows 0..63
    const int r1 = (tid + 256) >> 2;         // rows 64..127
    const int c8 = (tid & 3) * 8;            // 16B chunk within 32-k row
    const size_t gA0 = (size_t)(bm * 128 + r0) * 1024 + c8;
    const size_t gA1 = (size_t)(bm * 128 + r1) * 1024 + c8;
    const size_t gB0 = (size_t)(bn * 128 + r0) * 1024 + c8;
    const size_t gB1 = (size_t)(bn * 128 + r1) * 1024 + c8;

    auto issue_tile = [&](int buf, int k0) {
#if USE_ASYNC_LDS
        cp16_g2l(&As[buf][r0][c8], Ag + gA0 + k0);
        cp16_g2l(&As[buf][r1][c8], Ag + gA1 + k0);
        cp16_g2l(&Bs[buf][r0][c8], Bg + gB0 + k0);
        cp16_g2l(&Bs[buf][r1][c8], Bg + gB1 + k0);
#else
        const uint4 a0 = *(const uint4*)(Ag + gA0 + k0);
        const uint4 a1 = *(const uint4*)(Ag + gA1 + k0);
        const uint4 b0 = *(const uint4*)(Bg + gB0 + k0);
        const uint4 b1 = *(const uint4*)(Bg + gB1 + k0);
        *(uint4*)&As[buf][r0][c8] = a0;
        *(uint4*)&As[buf][r1][c8] = a1;
        *(uint4*)&Bs[buf][r0][c8] = b0;
        *(uint4*)&Bs[buf][r1][c8] = b1;
#endif
    };

    v8f acc[2][4] = {};
    issue_tile(0, 0);
    wait_async_lds();
    __syncthreads();

    for (int it = 0; it < 32; ++it) {
        const int buf = it & 1;
        if (it < 31) issue_tile(buf ^ 1, (it + 1) * 32);

        v16bf aF[2], bF[4];
#pragma unroll
        for (int mt = 0; mt < 2; ++mt)
            aF[mt] = frag_row<40>(&As[buf][0][0], wm * 32 + mt * 16, 0, lane);
#pragma unroll
        for (int nt = 0; nt < 4; ++nt)
            bF[nt] = frag_row<40>(&Bs[buf][0][0], wn * 64 + nt * 16, 0, lane);
#pragma unroll
        for (int mt = 0; mt < 2; ++mt)
#pragma unroll
            for (int nt = 0; nt < 4; ++nt)
                acc[mt][nt] = wmma_bf16(aF[mt], bF[nt], acc[mt][nt]);

        if (it < 31) wait_async_lds();
        __syncthreads();
    }

#pragma unroll
    for (int mt = 0; mt < 2; ++mt)
#pragma unroll
        for (int nt = 0; nt < 4; ++nt)
#pragma unroll
            for (int r = 0; r < 8; ++r) {
                const int m = bm * 128 + wm * 32 + mt * 16 + r + (lane >> 4) * 8;
                const int nn = bn * 128 + wn * 64 + nt * 16 + (lane & 15);
                float v = acc[mt][nt][r];
                v += ALPHA_C * img[(size_t)m * 1024 + (nn & 1023)];
                if (nn >= 2048)
                    v *= (1.0f - exp2f(-5.0f - (float)((nn >> 6) & 15)));  // gamma
                qkv[(size_t)m * 3072 + nn] = v;
            }
}

// ---------------------------------------------------------------------------
// vsum[b,n,w,d] = sum_h v[b,n,h,w,d]
// ---------------------------------------------------------------------------
__global__ __launch_bounds__(256) void vsum_kernel(const float* __restrict__ qkv,
                                                   float* __restrict__ vs) {
    const int idx = blockIdx.x * 256 + threadIdx.x;
    int t = idx;
    const int d = t & 63; t >>= 6;
    const int w = t % 28; t /= 28;
    const int n = t & 15; const int b = t >> 4;
    float s = 0.f;
    for (int h = 0; h < 28; ++h)
        s += qkv[((size_t)((b * 28 + h) * 28 + w)) * 3072 + 1024 + n * 64 + d];
    vs[idx] = s;
}

// ---------------------------------------------------------------------------
// attn: per (b,n,h): RoPE(q,k) -> scores = q k^T -> out = scores @ vsum (WMMA)
// ---------------------------------------------------------------------------
__global__ __launch_bounds__(256)
void attn_kernel(const float* __restrict__ qkv, const float* __restrict__ vs,
                 float* __restrict__ attn_out, float* __restrict__ stats) {
    __shared__ __bf16 Qs[32][64];
    __shared__ __bf16 Ks[32][64];
    __shared__ __bf16 Vt[64][40];   // transposed vsum: Vt[d][y]
    __shared__ __bf16 Ss[32][40];
    __shared__ float  red[256];

    const int bid = blockIdx.x;
    const int h = bid % 28;
    const int n = (bid / 28) & 15;
    const int b = bid / (28 * 16);
    const int tid = threadIdx.x;
    const int lane = tid & 31, wave = tid >> 5;

    for (int i = tid; i < 1024; i += 256) {   // 32 w * 2 halves * 16 j
        const int j = i & 15;
        const int half = (i >> 4) & 1;
        const int w = i >> 5;
        const int p0 = half * 32 + j, p1 = p0 + 16;
        if (w < 28) {
            const size_t m = (size_t)(b * 28 + h) * 28 + w;
            const float* qb = qkv + m * 3072 + n * 64;
            const float q0 = qb[p0], q1 = qb[p1];
            const float kk0 = qb[2048 + p0], kk1 = qb[2048 + p1];
            const float pos = half ? (float)w : (float)h;
            const float ang = pos * __powf(10000.0f, -(float)(2 * j) * (1.0f / 32.0f));
            float sn, cs;
            __sincosf(ang, &sn, &cs);
            Qs[w][p0] = (__bf16)(q0 * cs - q1 * sn);
            Qs[w][p1] = (__bf16)(q1 * cs + q0 * sn);
            Ks[w][p0] = (__bf16)(kk0 * cs - kk1 * sn);
            Ks[w][p1] = (__bf16)(kk1 * cs + kk0 * sn);
        } else {
            Qs[w][p0] = (__bf16)0.f; Qs[w][p1] = (__bf16)0.f;
            Ks[w][p0] = (__bf16)0.f; Ks[w][p1] = (__bf16)0.f;
        }
    }
    for (int i = tid; i < 2048; i += 256) {   // Vt[d][y], zero-pad y 28..31
        const int w = i >> 6, d = i & 63;
        const float v = (w < 28) ? vs[((size_t)(b * 16 + n) * 28 + w) * 64 + d] : 0.f;
        Vt[d][w] = (__bf16)v;
    }
    __syncthreads();

    if (wave < 4) {                            // scores, 2x2 tiles, K=64
        const int mt = wave >> 1, nt = wave & 1;
        v8f sacc = {};
#pragma unroll
        for (int ks = 0; ks < 64; ks += 32) {
            const v16bf a  = frag_row<64>(&Qs[0][0], mt * 16, ks, lane);
            const v16bf bb = frag_row<64>(&Ks[0][0], nt * 16, ks, lane);  // B = K^T
            sacc = wmma_bf16(a, bb, sacc);
        }
#pragma unroll
        for (int r = 0; r < 8; ++r)
            Ss[mt * 16 + r + (lane >> 4) * 8][nt * 16 + (lane & 15)] = (__bf16)sacc[r];
    }
    __syncthreads();

    const int mt = wave >> 2, nt = wave & 3;   // out, 2x4 tiles, K=32
    const v16bf a  = frag_row<40>(&Ss[0][0], mt * 16, 0, lane);
    const v16bf bb = frag_row<40>(&Vt[0][0], nt * 16, 0, lane);
    v8f oacc = {};
    oacc = wmma_bf16(a, bb, oacc);

    float s1 = 0.f, s2 = 0.f;
#pragma unroll
    for (int r = 0; r < 8; ++r) {
        const int w = mt * 16 + r + (lane >> 4) * 8;
        const int d = nt * 16 + (lane & 15);
        const float v = oacc[r];
        if (w < 28) {
            attn_out[(((size_t)(b * 16 + n) * 28 + h) * 28 + w) * 64 + d] = v;
            s1 += v; s2 += v * v;
        }
    }
    red[tid] = s1; __syncthreads();
    for (int off = 128; off > 0; off >>= 1) {
        if (tid < off) red[tid] += red[tid + off];
        __syncthreads();
    }
    if (tid == 0) atomicAdd(&stats[2 * (b * 16 + n)], red[0]);
    __syncthreads();
    red[tid] = s2; __syncthreads();
    for (int off = 128; off > 0; off >>= 1) {
        if (tid < off) red[tid] += red[tid + off];
        __syncthreads();
    }
    if (tid == 0) atomicAdd(&stats[2 * (b * 16 + n) + 1], red[0]);
}

__global__ __launch_bounds__(256)
void stats_finalize_kernel(const float* __restrict__ stats, const float* __restrict__ gn_w,
                           const float* __restrict__ gn_b, float* __restrict__ sshift) {
    const int g = blockIdx.x * 256 + threadIdx.x;
    if (g < 512) {
        const float cnt = 28.0f * 28.0f * 64.0f;
        const float mu = stats[2 * g] / cnt;
        const float var = stats[2 * g + 1] / cnt - mu * mu;
        const float rs = rsqrtf(var + EPS_C);
        const int hd = g & 15;
        const float sc = gn_w[hd] * rs;
        sshift[2 * g]     = sc;
        sshift[2 * g + 1] = gn_b[hd] - mu * sc;
    }
}

// GroupNorm-fused f32 -> bf16 (layout kept (b,n,x,w,d); 50176 elems per group)
__global__ __launch_bounds__(256)
void normalize_kernel(const float* __restrict__ attn, const float* __restrict__ ss,
                      __bf16* __restrict__ nrm) {
    const size_t i = ((size_t)blockIdx.x * 256 + threadIdx.x) * 4;
    const int g = (int)(i / 50176);
    const float sc = ss[2 * g], sh = ss[2 * g + 1];
    const float4 v = *(const float4*)(attn + i);
    nrm[i + 0] = (__bf16)(v.x * sc + sh);
    nrm[i + 1] = (__bf16)(v.y * sc + sh);
    nrm[i + 2] = (__bf16)(v.z * sc + sh);
    nrm[i + 3] = (__bf16)(v.w * sc + sh);
}

// ---------------------------------------------------------------------------
// GEMM2: out = nrm(permuted) @ o_w  (M=25088,K=1024,N=1024)
// ---------------------------------------------------------------------------
__global__ __launch_bounds__(256)
void gemm_out_kernel(const __bf16* __restrict__ nrm, const __bf16* __restrict__ Bg,
                     float* __restrict__ out) {
    __shared__ __bf16 As[2][128][40];
    __shared__ __bf16 Bs[2][128][40];
    const int bn = blockIdx.x;        // 0..7
    const int bm = blockIdx.y;        // 0..195
    const int tid = threadIdx.x;
    const int lane = tid & 31, wave = tid >> 5;
    const int wm = wave & 3, wn = wave >> 2;

    const int r0 = tid >> 2;
    const int r1 = (tid + 256) >> 2;
    const int c8 = (tid & 3) * 8;
    const size_t gB0 = (size_t)(bn * 128 + r0) * 1024 + c8;
    const size_t gB1 = (size_t)(bn * 128 + r1) * 1024 + c8;

    auto a_src = [&](int row, int k0) -> const __bf16* {
        const int m = bm * 128 + row;
        const int b_ = m / 784, rem = m % 784;
        const int xx = rem / 28, w = rem % 28;
        const int head = k0 >> 6, dbase = k0 & 63;
        return nrm + (((size_t)(b_ * 16 + head) * 28 + xx) * 28 + w) * 64 + dbase + c8;
    };

    auto issue_tile = [&](int buf, int k0) {
#if USE_ASYNC_LDS
        cp16_g2l(&As[buf][r0][c8], a_src(r0, k0));
        cp16_g2l(&As[buf][r1][c8], a_src(r1, k0));
        cp16_g2l(&Bs[buf][r0][c8], Bg + gB0 + k0);
        cp16_g2l(&Bs[buf][r1][c8], Bg + gB1 + k0);
#else
        const uint4 a0 = *(const uint4*)a_src(r0, k0);
        const uint4 a1 = *(const uint4*)a_src(r1, k0);
        const uint4 b0 = *(const uint4*)(Bg + gB0 + k0);
        const uint4 b1 = *(const uint4*)(Bg + gB1 + k0);
        *(uint4*)&As[buf][r0][c8] = a0;
        *(uint4*)&As[buf][r1][c8] = a1;
        *(uint4*)&Bs[buf][r0][c8] = b0;
        *(uint4*)&Bs[buf][r1][c8] = b1;
#endif
    };

    v8f acc[2][4] = {};
    issue_tile(0, 0);
    wait_async_lds();
    __syncthreads();

    for (int it = 0; it < 32; ++it) {
        const int buf = it & 1;
        if (it < 31) issue_tile(buf ^ 1, (it + 1) * 32);

        v16bf aF[2], bF[4];
#pragma unroll
        for (int mt = 0; mt < 2; ++mt)
            aF[mt] = frag_row<40>(&As[buf][0][0], wm * 32 + mt * 16, 0, lane);
#pragma unroll
        for (int nt = 0; nt < 4; ++nt)
            bF[nt] = frag_row<40>(&Bs[buf][0][0], wn * 64 + nt * 16, 0, lane);
#pragma unroll
        for (int mt = 0; mt < 2; ++mt)
#pragma unroll
            for (int nt = 0; nt < 4; ++nt)
                acc[mt][nt] = wmma_bf16(aF[mt], bF[nt], acc[mt][nt]);

        if (it < 31) wait_async_lds();
        __syncthreads();
    }

#pragma unroll
    for (int mt = 0; mt < 2; ++mt)
#pragma unroll
        for (int nt = 0; nt < 4; ++nt)
#pragma unroll
            for (int r = 0; r < 8; ++r) {
                const int m = bm * 128 + wm * 32 + mt * 16 + r + (lane >> 4) * 8;
                const int nn = bn * 128 + wn * 64 + nt * 16 + (lane & 15);
                out[(size_t)m * 1024 + nn] = acc[mt][nt][r];
            }
}

// ---------------------------------------------------------------------------
extern "C" void kernel_launch(void* const* d_in, const int* in_sizes, int n_in,
                              void* d_out, int out_size, void* d_ws, size_t ws_size,
                              hipStream_t stream) {
    const float* x     = (const float*)d_in[0];
    const float* img   = (const float*)d_in[1];
    const float* qkv_w = (const float*)d_in[2];   // (1024,3072)
    const float* o_w   = (const float*)d_in[3];   // (1024,1024)
    const float* gn_w  = (const float*)d_in[4];
    const float* gn_b  = (const float*)d_in[5];
    float* out = (float*)d_out;

    char* base = (char*)d_ws;
    float*  qkv    = (float*)(base);                      // 308,281,344 B
    float*  attn   = (float*)(base + 308281344ull);       // 102,760,448 B
    float*  vs     = (float*)(base + 411041792ull);       //   3,670,016 B
    float*  stats  = (float*)(base + 414711808ull);       //       4,096 B
    float*  sshift = (float*)(base + 414715904ull);       //       4,096 B
    __bf16* xb     = (__bf16*)(base + 414720000ull);      //  51,380,224 B
    __bf16* wqkvT  = (__bf16*)(base + 466100224ull);      //   6,291,456 B
    __bf16* woT    = (__bf16*)(base + 472391680ull);      //   2,097,152 B
    __bf16* nrm    = (__bf16*)(base + 474488832ull);      //  51,380,224 B

    cvt_bf16_kernel<<<25088, 256, 0, stream>>>(x, xb);
    transpose_bf16_kernel<<<12288, 256, 0, stream>>>(qkv_w, wqkvT, 1024, 3072);
    transpose_bf16_kernel<<<4096, 256, 0, stream>>>(o_w, woT, 1024, 1024);

    gemm_qkv_kernel<<<dim3(24, 196), 256, 0, stream>>>(xb, wqkvT, img, qkv);
    vsum_kernel<<<3584, 256, 0, stream>>>(qkv, vs);
    zero_kernel<<<4, 256, 0, stream>>>(stats, 1024);
    attn_kernel<<<32 * 16 * 28, 256, 0, stream>>>(qkv, vs, attn, stats);
    stats_finalize_kernel<<<2, 256, 0, stream>>>(stats, gn_w, gn_b, sshift);
    normalize_kernel<<<25088, 256, 0, stream>>>(attn, sshift, nrm);
    gemm_out_kernel<<<dim3(8, 196), 256, 0, stream>>>(nrm, woT, out);
}
